// LlamaQForCausalLM_21947282883015
// MI455X (gfx1250) — matrix operand order repaired and checked by reading, not compiled, not run on hardware
//
#include <hip/hip_runtime.h>
#include <hip/hip_bf16.h>

// ---------------- model constants ----------------
#define LNUM   4
#define H      2048
#define NH     16
#define HD     128
#define INTER  5632
#define VOCAB  32000
#define GRP    128
#define BB     2
#define SS     1024
#define MTOT   (BB * SS)        // 2048 token rows
#define GQ     (H / GRP)        // 16
#define GI     (INTER / GRP)    // 44
#define EPS    1e-6f
#define ATT_SCALE 0.08838834764831845f  // HD^-0.5

typedef __bf16 bf16;
typedef __attribute__((ext_vector_type(16))) __bf16 v16bf;
typedef __attribute__((ext_vector_type(8)))  __bf16 v8bf;
typedef __attribute__((ext_vector_type(8)))  float  v8f;

// ---------------- WMMA helpers ----------------
__device__ __forceinline__ v8f wmma_bf16(v16bf a, v16bf b, v8f c) {
  // D = A(16x32 bf16) * B(32x16 bf16) + C(16x16 f32)
  return __builtin_amdgcn_wmma_f32_16x16x32_bf16(false, a, false, b, (short)0, c,
                                                 false, false);
}

// A fragment: row-major A[m0..m0+15][k0..k0+31].
// Layout: lanes 0-15 hold row M=lane, K 0-15;
//         lanes 16-31 hold row M=lane-16, K 16-31.
__device__ __forceinline__ v16bf load_a_frag(const bf16* __restrict__ A, int lda,
                                             int m0, int k0) {
  int lane = threadIdx.x & 31;
  const bf16* p = A + (size_t)(m0 + (lane & 15)) * lda + k0 + ((lane >> 4) ? 16 : 0);
  v8bf lo = *(const v8bf*)p;         // K = koff + 0..7
  v8bf hi = *(const v8bf*)(p + 8);   // K = koff + 8..15
  v16bf f;
#pragma unroll
  for (int i = 0; i < 8; ++i) { f[i] = lo[i]; f[i + 8] = hi[i]; }
  return f;
}

// B fragment from a transposed layout Bt[n][k] (contiguous k).
// Layout used here (contiguous k, same as A): lane holds column N=lane&15; lanes 0-15 K=0..15, lanes 16-31 K=16..31.
__device__ __forceinline__ v16bf load_bt_frag(const bf16* __restrict__ Bt, int ldb,
                                              int n0, int k0) {
  int lane = threadIdx.x & 31;
  const bf16* p = Bt + (size_t)(n0 + (lane & 15)) * ldb + k0 + ((lane >> 4) ? 16 : 0);
  return *(const v16bf*)p;           // 32B contiguous
}

// B fragment from fp32 transposed weights (lm_head: W[v][h] row-major == Bt).
__device__ __forceinline__ v16bf load_bt_frag_f32(const float* __restrict__ Bt, int ldb,
                                                  int n0, int k0) {
  int lane = threadIdx.x & 31;
  const float* p = Bt + (size_t)(n0 + (lane & 15)) * ldb + k0 + ((lane >> 4) ? 16 : 0);
  v16bf f;
#pragma unroll
  for (int i = 0; i < 16; ++i) f[i] = (bf16)p[i];
  return f;
}

// C/D tile: VGPR r, lanes 0-15 -> row r, col lane; lanes 16-31 -> row 8+r.
__device__ __forceinline__ void store_c_f32(float* __restrict__ C, int ldc,
                                            int m0, int n0, v8f acc) {
  int lane = threadIdx.x & 31;
  int col = n0 + (lane & 15);
  int rb  = m0 + ((lane >> 4) ? 8 : 0);
#pragma unroll
  for (int r = 0; r < 8; ++r) C[(size_t)(rb + r) * ldc + col] = acc[r];
}

// ---------------- kernels ----------------
__global__ __launch_bounds__(256) void embed_gather_kernel(
    const int* __restrict__ ids, const float* __restrict__ E, float* __restrict__ X) {
  int row = blockIdx.x;
  int id = ids[row];
  const float* src = E + (size_t)id * H;
  float* dst = X + (size_t)row * H;
  for (int i = threadIdx.x; i < H; i += 256) dst[i] = src[i];
}

__global__ __launch_bounds__(256) void rmsnorm_bf16_kernel(
    const float* __restrict__ X, const float* __restrict__ w, bf16* __restrict__ Y) {
  int row = blockIdx.x;
  const float* x = X + (size_t)row * H;
  __shared__ float red[8];
  __shared__ float srstd;
  float ss = 0.f;
  for (int i = threadIdx.x; i < H; i += 256) { float v = x[i]; ss += v * v; }
#pragma unroll
  for (int o = 16; o > 0; o >>= 1) ss += __shfl_xor(ss, o, 32);
  if ((threadIdx.x & 31) == 0) red[threadIdx.x >> 5] = ss;
  __syncthreads();
  if (threadIdx.x == 0) {
    float t = 0.f;
#pragma unroll
    for (int i = 0; i < 8; ++i) t += red[i];
    srstd = rsqrtf(t / (float)H + EPS);
  }
  __syncthreads();
  float r = srstd;
  bf16* y = Y + (size_t)row * H;
  for (int i = threadIdx.x; i < H; i += 256) y[i] = (bf16)(x[i] * r * w[i]);
}

// C[M,N] = A[M,K](bf16) x dequant(qw[K,N] int4-in-int32, per-128-group zero/scale)
// Block tile: 256(M) x 64(N), 8 waves: each wave owns a 32-row strip (2 A frags,
// 8 accumulators). K staged 64 wide per barrier (two WMMA K-steps).
__global__ __launch_bounds__(256) void gemm_awq_kernel(
    const bf16* __restrict__ A, const int* __restrict__ qw, const int* __restrict__ qz,
    const float* __restrict__ sc, float* __restrict__ C, int N, int K) {
  __shared__ __align__(32) bf16 Bs[64][80];   // transposed [n][k], 64 k + pad
  int n0 = blockIdx.x * 64;
  int m0 = blockIdx.y * 256;
  int tid = threadIdx.x;
  int wave = tid >> 5;
  int wm = m0 + wave * 32;
  // per-thread dequant coordinates: 64k x 64n tile, 16 consecutive n per thread
  int dq_k = tid >> 2;            // 0..63
  int dq_n = (tid & 3) * 16;      // 0,16,32,48
  v8f acc[2][4] = {};
  for (int k0 = 0; k0 < K; k0 += 64) {
    int g = k0 >> 7;  // 64 | k0  =>  whole K-step inside one quant group
    const int*   qwp = qw + (size_t)(k0 + dq_k) * N + n0 + dq_n;
    const int*   qzp = qz + (size_t)g * N + n0 + dq_n;
    const float* scp = sc + (size_t)g * N + n0 + dq_n;
    if (k0 + 64 < K)  // prefetch next K-block of weights (global_prefetch)
      __builtin_prefetch(qw + (size_t)(k0 + 64 + dq_k) * N + n0 + dq_n, 0, 1);
#pragma unroll
    for (int i = 0; i < 16; ++i) {
      float wv = (float)qwp[i];
      float zv = (float)qzp[i];
      float sv = scp[i];
      Bs[dq_n + i][dq_k] = (bf16)((wv - zv) * sv);
    }
    __syncthreads();
#pragma unroll
    for (int ks = 0; ks < 2; ++ks) {
      v16bf a0 = load_a_frag(A, K, wm, k0 + ks * 32);
      v16bf a1 = load_a_frag(A, K, wm + 16, k0 + ks * 32);
#pragma unroll
      for (int t = 0; t < 4; ++t) {
        v16bf b = load_bt_frag(&Bs[0][0], 80, t * 16, ks * 32);
        acc[0][t] = wmma_bf16(a0, b, acc[0][t]);
        acc[1][t] = wmma_bf16(a1, b, acc[1][t]);
      }
    }
    __syncthreads();
  }
#pragma unroll
  for (int mi = 0; mi < 2; ++mi)
#pragma unroll
    for (int t = 0; t < 4; ++t)
      store_c_f32(C, N, wm + mi * 16, n0 + t * 16, acc[mi][t]);
}

// QKV[B,S,3H] fp32 -> RoPE'd Qb,Kb [b,h,s,d] bf16 ; V transposed Vt [b,h,d,s] bf16
__global__ __launch_bounds__(256) void qkv_rope_kernel(
    const float* __restrict__ QKV, const int* __restrict__ pos,
    bf16* __restrict__ Qb, bf16* __restrict__ Kb, bf16* __restrict__ Vt) {
  int bs = blockIdx.x;
  int b = bs / SS, s = bs % SS;
  const float* p = QKV + (size_t)bs * (3 * H);
  float ps = (float)pos[bs];
  for (int j = threadIdx.x; j < NH * 64; j += 256) {
    int h = j >> 6, dh = j & 63;
    float inv_freq = __expf(-0.14391156831f * (float)dh);  // 10000^(-dh/64)
    float ang = ps * inv_freq;
    float c = __cosf(ang), sn = __sinf(ang);
    size_t base = ((size_t)(b * NH + h) * SS + s) * HD;
    float q1 = p[h * HD + dh], q2 = p[h * HD + dh + 64];
    Qb[base + dh]      = (bf16)(q1 * c - q2 * sn);
    Qb[base + dh + 64] = (bf16)(q2 * c + q1 * sn);
    float k1 = p[H + h * HD + dh], k2 = p[H + h * HD + dh + 64];
    Kb[base + dh]      = (bf16)(k1 * c - k2 * sn);
    Kb[base + dh + 64] = (bf16)(k2 * c + k1 * sn);
  }
  for (int j = threadIdx.x; j < H; j += 256) {
    int h = j >> 7, d = j & 127;
    Vt[((size_t)(b * NH + h) * HD + d) * SS + s] = (bf16)p[2 * H + j];
  }
}

// One workgroup (4 waves) per (b, h, 16-query tile). Causal attention.
__global__ __launch_bounds__(128) void attn_kernel(
    const bf16* __restrict__ Qb, const bf16* __restrict__ Kb,
    const bf16* __restrict__ Vt, bf16* __restrict__ O) {
  __shared__ __align__(32) bf16 sm[16][1040];   // scores/probs, row-major [q][s]
  int blk = blockIdx.x;
  int qt = blk & 63;           // S/16 = 64 tiles
  int bh = blk >> 6;           // b*NH + h
  int q0 = qt * 16;
  int L  = q0 + 16;            // valid key columns
  int Lp = (L + 31) & ~31;     // padded to WMMA K multiple
  const bf16* Qh = Qb + (size_t)bh * SS * HD;
  const bf16* Kh = Kb + (size_t)bh * SS * HD;
  const bf16* Vh = Vt + (size_t)bh * HD * SS;
  int wave = threadIdx.x >> 5, lane = threadIdx.x & 31;

  // Q fragments (16 x 128) held across the whole kernel
  v16bf qf[4];
#pragma unroll
  for (int t = 0; t < 4; ++t) qf[t] = load_a_frag(Qh, HD, q0, t * 32);

  // ---- scores = scale * Q K^T, causal-masked, into LDS (bf16) ----
  for (int s0 = wave * 16; s0 < L; s0 += 64) {
    v8f acc = {};
#pragma unroll
    for (int t = 0; t < 4; ++t) {
      v16bf kf = load_bt_frag(Kh, HD, s0, t * 32);
      acc = wmma_bf16(qf[t], kf, acc);
    }
    int col = s0 + (lane & 15);
    int rb  = (lane >> 4) ? 8 : 0;
#pragma unroll
    for (int r = 0; r < 8; ++r) {
      int row = rb + r;
      float v = acc[r] * ATT_SCALE;
      if (col > q0 + row) v = -1e30f;
      sm[row][col] = (bf16)v;
    }
  }
  __syncthreads();

  // ---- softmax: 16 rows x 8 lanes/row ----
  {
    int row = threadIdx.x >> 3;
    int ch  = threadIdx.x & 7;
    float m = -1e30f;
    for (int j = ch; j < L; j += 8) m = fmaxf(m, (float)sm[row][j]);
#pragma unroll
    for (int o = 1; o < 8; o <<= 1) m = fmaxf(m, __shfl_xor(m, o, 32));
    float sum = 0.f;
    for (int j = ch; j < L; j += 8) {
      float e = __expf((float)sm[row][j] - m);
      sm[row][j] = (bf16)e;
      sum += e;
    }
#pragma unroll
    for (int o = 1; o < 8; o <<= 1) sum += __shfl_xor(sum, o, 32);
    float inv = 1.0f / sum;
    for (int j = ch; j < L; j += 8) sm[row][j] = (bf16)((float)sm[row][j] * inv);
    for (int j = L + ch; j < Lp; j += 8) sm[row][j] = (bf16)0.0f;  // zero pad tile
  }
  __syncthreads();

  // ---- out = probs @ V : wave w handles d-range [w*32, w*32+32) ----
  v8f oacc[2] = {v8f{}, v8f{}};
  const bf16* smp = &sm[0][0];
  for (int s0 = 0; s0 < Lp; s0 += 32) {
    v16bf pa = load_a_frag(smp, 1040, 0, s0);
#pragma unroll
    for (int t = 0; t < 2; ++t) {
      v16bf vf = load_bt_frag(Vh, SS, wave * 32 + t * 16, s0);
      oacc[t] = wmma_bf16(pa, vf, oacc[t]);
    }
  }
  int b = bh >> 4, h = bh & 15;
#pragma unroll
  for (int t = 0; t < 2; ++t) {
    int d  = wave * 32 + t * 16 + (lane & 15);
    int rb = (lane >> 4) ? 8 : 0;
#pragma unroll
    for (int r = 0; r < 8; ++r) {
      int s = q0 + rb + r;
      O[(((size_t)b * SS + s) * NH + h) * HD + d] = (bf16)oacc[t][r];
    }
  }
}

__global__ __launch_bounds__(256) void add_residual_kernel(
    float* __restrict__ X, const float* __restrict__ D, int n) {
  int i = blockIdx.x * 256 + threadIdx.x;
  if (i < n) X[i] += D[i];
}

__global__ __launch_bounds__(256) void silu_mul_kernel(
    const float* __restrict__ GU, bf16* __restrict__ Hm, int n) {
  int i = blockIdx.x * 256 + threadIdx.x;
  if (i >= n) return;
  int m = i / INTER, j = i - m * INTER;
  float g = GU[(size_t)m * (2 * INTER) + j];
  float u = GU[(size_t)m * (2 * INTER) + INTER + j];
  float sil = g / (1.0f + __expf(-g));
  Hm[i] = (bf16)(sil * u);
}

// logits[M, V] = A[M, H](bf16) x W[V, H]^T (fp32, converted per-fragment)
// 256(M) x 64(N) block tile, 32 rows per wave.
__global__ __launch_bounds__(256) void gemm_lmhead_kernel(
    const bf16* __restrict__ A, const float* __restrict__ W, float* __restrict__ C) {
  int n0 = blockIdx.x * 64;
  int m0 = blockIdx.y * 256;
  int wave = threadIdx.x >> 5;
  int wm = m0 + wave * 32;
  v8f acc[2][4] = {};
  for (int k0 = 0; k0 < H; k0 += 32) {
    v16bf a0 = load_a_frag(A, H, wm, k0);
    v16bf a1 = load_a_frag(A, H, wm + 16, k0);
#pragma unroll
    for (int t = 0; t < 4; ++t) {
      v16bf b = load_bt_frag_f32(W, H, n0 + t * 16, k0);
      acc[0][t] = wmma_bf16(a0, b, acc[0][t]);
      acc[1][t] = wmma_bf16(a1, b, acc[1][t]);
    }
  }
#pragma unroll
  for (int mi = 0; mi < 2; ++mi)
#pragma unroll
    for (int t = 0; t < 4; ++t)
      store_c_f32(C, VOCAB, wm + mi * 16, n0 + t * 16, acc[mi][t]);
}

// ---------------- host launcher ----------------
extern "C" void kernel_launch(void* const* d_in, const int* in_sizes, int n_in,
                              void* d_out, int out_size, void* d_ws, size_t ws_size,
                              hipStream_t stream) {
  (void)in_sizes; (void)n_in; (void)out_size; (void)ws_size;
  const int*   ids     = (const int*)d_in[0];
  const int*   pos     = (const int*)d_in[1];
  const float* embed_w = (const float*)d_in[2];
  const int*   qkv_qw  = (const int*)d_in[3];
  const int*   qkv_qz  = (const int*)d_in[4];
  const float* qkv_sc  = (const float*)d_in[5];
  const int*   o_qw    = (const int*)d_in[6];
  const int*   o_qz    = (const int*)d_in[7];
  const float* o_sc    = (const float*)d_in[8];
  const int*   gu_qw   = (const int*)d_in[9];
  const int*   gu_qz   = (const int*)d_in[10];
  const float* gu_sc   = (const float*)d_in[11];
  const int*   dn_qw   = (const int*)d_in[12];
  const int*   dn_qz   = (const int*)d_in[13];
  const float* dn_sc   = (const float*)d_in[14];
  const float* ln1_w   = (const float*)d_in[15];
  const float* ln2_w   = (const float*)d_in[16];
  const float* norm_w  = (const float*)d_in[17];
  const float* lm_head = (const float*)d_in[18];
  float* out = (float*)d_out;

  const size_t MB = 1024ull * 1024ull;
  char* ws = (char*)d_ws;
  float* X   = (float*)(ws + 0);          // residual fp32 [2048, 2048]   (16 MB)
  bf16*  XN  = (bf16*)(ws + 16 * MB);     // normed acts bf16             (8 MB)
  float* BIG = (float*)(ws + 24 * MB);    // gemm out fp32, max 2048x11264 (88 MB)
  bf16*  QB  = (bf16*)(ws + 112 * MB);    // [b,h,s,d] bf16               (8 MB)
  bf16*  KB  = (bf16*)(ws + 120 * MB);
  bf16*  VT  = (bf16*)(ws + 128 * MB);    // [b,h,d,s]
  bf16*  ATT = (bf16*)(ws + 136 * MB);    // attn out bf16 [b,s,h,d]      (8 MB)
  bf16*  HM  = (bf16*)(ws + 144 * MB);    // silu*up bf16 [2048, 5632]    (22 MB)

  embed_gather_kernel<<<MTOT, 256, 0, stream>>>(ids, embed_w, X);

  for (int l = 0; l < LNUM; ++l) {
    // attention block
    rmsnorm_bf16_kernel<<<MTOT, 256, 0, stream>>>(X, ln1_w + (size_t)l * H, XN);
    gemm_awq_kernel<<<dim3((3 * H) / 64, MTOT / 256), 256, 0, stream>>>(
        XN, qkv_qw + (size_t)l * H * 3 * H, qkv_qz + (size_t)l * GQ * 3 * H,
        qkv_sc + (size_t)l * GQ * 3 * H, BIG, 3 * H, H);
    qkv_rope_kernel<<<MTOT, 256, 0, stream>>>(BIG, pos, QB, KB, VT);
    attn_kernel<<<BB * NH * (SS / 16), 128, 0, stream>>>(QB, KB, VT, ATT);
    gemm_awq_kernel<<<dim3(H / 64, MTOT / 256), 256, 0, stream>>>(
        ATT, o_qw + (size_t)l * H * H, o_qz + (size_t)l * GQ * H,
        o_sc + (size_t)l * GQ * H, BIG, H, H);
    add_residual_kernel<<<(MTOT * H) / 256, 256, 0, stream>>>(X, BIG, MTOT * H);

    // MLP block
    rmsnorm_bf16_kernel<<<MTOT, 256, 0, stream>>>(X, ln2_w + (size_t)l * H, XN);
    gemm_awq_kernel<<<dim3((2 * INTER) / 64, MTOT / 256), 256, 0, stream>>>(
        XN, gu_qw + (size_t)l * H * 2 * INTER, gu_qz + (size_t)l * GQ * 2 * INTER,
        gu_sc + (size_t)l * GQ * 2 * INTER, BIG, 2 * INTER, H);
    silu_mul_kernel<<<(MTOT * INTER + 255) / 256, 256, 0, stream>>>(BIG, HM, MTOT * INTER);
    gemm_awq_kernel<<<dim3(H / 64, MTOT / 256), 256, 0, stream>>>(
        HM, dn_qw + (size_t)l * INTER * H, dn_qz + (size_t)l * GI * H,
        dn_sc + (size_t)l * GI * H, BIG, H, INTER);
    add_residual_kernel<<<(MTOT * H) / 256, 256, 0, stream>>>(X, BIG, MTOT * H);
  }

  rmsnorm_bf16_kernel<<<MTOT, 256, 0, stream>>>(X, norm_w, XN);
  gemm_lmhead_kernel<<<dim3(VOCAB / 64, MTOT / 256), 256, 0, stream>>>(XN, lm_head, out);
}
